// GNN_Model_47648367182534
// MI455X (gfx1250) — compile-verified, matrix operand order checked by reading, not run on hardware
//
#include <hip/hip_runtime.h>
#include <stdint.h>

// ---------------- problem constants (from reference) ----------------
constexpr int J_NUM = 2048;
constexpr int C_NUM = 32;
constexpr int NN    = J_NUM + C_NUM;   // 2080 nodes
constexpr int HID   = 256;
constexpr int HEADS = 8;
constexpr int CH    = 32;
constexpr int E2    = J_NUM * C_NUM;   // 65536 decoder edges
constexpr int E1    = 2 * E2;          // 131072 message edges
constexpr int ETOT  = E1 + NN;         // + self loops = 133152
constexpr float NEG_SLOPE = 0.2f;

// ---------------- WMMA types ----------------
typedef __attribute__((ext_vector_type(16))) __bf16 v16bf;
typedef __attribute__((ext_vector_type(8)))  __bf16 bf16x8;
typedef __attribute__((ext_vector_type(8)))  float  v8f;

union FragBF { v16bf v; bf16x8 h[2]; };

__device__ __forceinline__ unsigned short f32_to_bf16_bits(float f) {
    union { float f; unsigned int u; } x; x.f = f;
    unsigned int u = x.u;
    unsigned int r = u + 0x7FFFu + ((u >> 16) & 1u);   // round-to-nearest-even
    return (unsigned short)(r >> 16);
}
__device__ __forceinline__ float bf16_bits_to_f32(unsigned short b) {
    union { unsigned int u; float f; } x; x.u = ((unsigned int)b) << 16;
    return x.f;
}

// ---------------- encoder: x = [nf1@W1+b1 ; nf2@W2+b2] ----------------
__global__ void encoder_kernel(const float* __restrict__ nf1, const float* __restrict__ nf2,
                               const float* __restrict__ W1, const float* __restrict__ b1,
                               const float* __restrict__ W2, const float* __restrict__ b2,
                               float* __restrict__ x) {
    int node = blockIdx.x, f = threadIdx.x;
    float s;
    if (node < J_NUM) {
        s = b1[f];
        const float* r = nf1 + (size_t)node * 16;
        #pragma unroll
        for (int i = 0; i < 16; ++i) s += r[i] * W1[i * HID + f];
    } else {
        s = b2[f];
        const float* r = nf2 + (size_t)(node - J_NUM) * 8;
        #pragma unroll
        for (int i = 0; i < 8; ++i) s += r[i] * W2[i * HID + f];
    }
    x[(size_t)node * HID + f] = s;
}

// ---------------- helpers: f32->bf16, weight transpose ----------------
__global__ void to_bf16_kernel(const float* __restrict__ in, unsigned short* __restrict__ out, int n) {
    int i = blockIdx.x * 256 + threadIdx.x;
    if (i < n) out[i] = f32_to_bf16_bits(in[i]);
}
// in: [K,N] f32 row-major -> out: [N,K] bf16 row-major
__global__ void transpose_bf16_kernel(const float* __restrict__ in, unsigned short* __restrict__ out,
                                      int K, int N) {
    int idx = blockIdx.x * 256 + threadIdx.x;
    if (idx >= K * N) return;
    int n = idx / K, k = idx - n * K;
    out[(size_t)n * K + k] = f32_to_bf16_bits(in[(size_t)k * N + n]);
}

// ---------------- bf16 WMMA GEMM: out[M,N] = A[M,K] @ Bt[N,K]^T (+bias)(relu) ----------------
// block = 256 threads (8 waves); wave w -> 16 rows, 4 x 16-col accumulator tiles (16x64 strip)
template <bool RELU, bool OUTBF16>
__global__ void gemm_bf16_wmma(const unsigned short* __restrict__ A,
                               const unsigned short* __restrict__ Bt,
                               const float* __restrict__ bias,
                               void* __restrict__ out, int M, int K, int N) {
    const int wave = threadIdx.x >> 5;
    const int lane = threadIdx.x & 31;
    const int m0 = blockIdx.x * 128 + wave * 16;
    const int n0 = blockIdx.y * 64;
    const int mrow  = m0 + (lane & 15);
    const int mload = mrow < M ? mrow : 0;        // clamp OOB rows (stores guarded)
    const int khalf = (lane >> 4) * 8;            // ISA 16-bit A/B fragment layout

    v8f acc0 = {}, acc1 = {}, acc2 = {}, acc3 = {};
    const __bf16* Ab  = (const __bf16*)A;
    const __bf16* Btb = (const __bf16*)Bt;
    const int ncol = lane & 15;

    for (int kk = 0; kk < K; kk += 32) {
        FragBF a;
        const __bf16* ap = Ab + (size_t)mload * K + kk + khalf;
        a.h[0] = *(const bf16x8*)(ap);
        a.h[1] = *(const bf16x8*)(ap + 16);
        __builtin_prefetch(ap + 32, 0, 0);        // hint next k-block (global_prefetch)
        FragBF b0, b1, b2, b3;
        const __bf16* bp0 = Btb + (size_t)(n0 +  0 + ncol) * K + kk + khalf;
        const __bf16* bp1 = Btb + (size_t)(n0 + 16 + ncol) * K + kk + khalf;
        const __bf16* bp2 = Btb + (size_t)(n0 + 32 + ncol) * K + kk + khalf;
        const __bf16* bp3 = Btb + (size_t)(n0 + 48 + ncol) * K + kk + khalf;
        b0.h[0] = *(const bf16x8*)(bp0); b0.h[1] = *(const bf16x8*)(bp0 + 16);
        b1.h[0] = *(const bf16x8*)(bp1); b1.h[1] = *(const bf16x8*)(bp1 + 16);
        b2.h[0] = *(const bf16x8*)(bp2); b2.h[1] = *(const bf16x8*)(bp2 + 16);
        b3.h[0] = *(const bf16x8*)(bp3); b3.h[1] = *(const bf16x8*)(bp3 + 16);
        acc0 = __builtin_amdgcn_wmma_f32_16x16x32_bf16(false, a.v, false, b0.v, (short)0, acc0, false, false);
        acc1 = __builtin_amdgcn_wmma_f32_16x16x32_bf16(false, a.v, false, b1.v, (short)0, acc1, false, false);
        acc2 = __builtin_amdgcn_wmma_f32_16x16x32_bf16(false, a.v, false, b2.v, (short)0, acc2, false, false);
        acc3 = __builtin_amdgcn_wmma_f32_16x16x32_bf16(false, a.v, false, b3.v, (short)0, acc3, false, false);
    }

    v8f accs[4] = {acc0, acc1, acc2, acc3};
    #pragma unroll
    for (int t = 0; t < 4; ++t) {
        int n = n0 + t * 16 + ncol;
        float bv = bias ? bias[n] : 0.0f;
        #pragma unroll
        for (int r = 0; r < 8; ++r) {
            int m = m0 + (lane >> 4) * 8 + r;     // ISA f32 C/D layout
            if (m < M) {
                float v = accs[t][r] + bv;
                if (RELU) v = v > 0.f ? v : 0.f;
                if (OUTBF16) ((unsigned short*)out)[(size_t)m * N + n] = f32_to_bf16_bits(v);
                else         ((float*)out)[(size_t)m * N + n] = v;
            }
        }
    }
}

// ---------------- GATv2 attention logits per (edge, head) ----------------
__global__ void edge_logit_kernel(const float* __restrict__ xl, const float* __restrict__ xr,
                                  const float* __restrict__ att,
                                  const int* __restrict__ e1src, const int* __restrict__ e1dst,
                                  float* __restrict__ logit) {
    int idx = blockIdx.x * 256 + threadIdx.x;
    if (idx >= ETOT * HEADS) return;
    int e = idx >> 3, h = idx & 7;
    int src, dst;
    if (e < E1) { src = e1src[e]; dst = e1dst[e]; }
    else        { src = dst = e - E1; }            // self loops
    const float* pl = xl + (size_t)src * HID + h * CH;
    const float* pr = xr + (size_t)dst * HID + h * CH;
    const float* pa = att + h * CH;
    float s = 0.f;
    #pragma unroll
    for (int c = 0; c < CH; ++c) {
        float v = pl[c] + pr[c];
        v = v >= 0.f ? v : NEG_SLOPE * v;
        s += v * pa[c];
    }
    logit[idx] = s;
}

// ---------------- softmax+aggregate for job dst (33 in-edges), wave32 shuffles ----------------
__global__ void agg_job_kernel(const float* __restrict__ xold, const float* __restrict__ xl,
                               const float* __restrict__ logit, const float* __restrict__ bconv,
                               float* __restrict__ xnew) {
    int j = blockIdx.x;
    int h = threadIdx.x >> 5, lane = threadIdx.x & 31;
    // incoming: crane->job edges E2 + j*32 + i (src = crane i), plus self loop
    float lg  = logit[(size_t)(E2 + j * C_NUM + lane) * HEADS + h];
    float lgs = logit[(size_t)(E1 + j) * HEADS + h];
    float mx = lg;
    for (int o = 16; o >= 1; o >>= 1) mx = fmaxf(mx, __shfl_xor(mx, o, 32));
    mx = fmaxf(mx, lgs);
    float ex = __expf(lg - lgs * 0.f - mx);
    float sum = ex;
    for (int o = 16; o >= 1; o >>= 1) sum += __shfl_xor(sum, o, 32);
    float exs = __expf(lgs - mx);
    float denom = sum + exs;
    float alpha = ex / denom;
    size_t fo = (size_t)h * CH + lane;            // lane == channel
    float acc = (exs / denom) * xl[(size_t)j * HID + fo];
    #pragma unroll 4
    for (int i = 0; i < C_NUM; ++i) {
        float ai = __shfl(alpha, i, 32);
        acc += ai * xl[(size_t)(J_NUM + i) * HID + fo];
    }
    size_t o = (size_t)j * HID + fo;
    xnew[o] = xold[o] + acc + bconv[fo];
}

// ---------------- softmax+aggregate for crane dst (2049 in-edges), block reduce ----------------
__global__ void agg_crane_kernel(const float* __restrict__ xold, const float* __restrict__ xl,
                                 const float* __restrict__ logit, const float* __restrict__ bconv,
                                 float* __restrict__ xnew) {
    __shared__ float red[256];
    int c = blockIdx.x >> 3, h = blockIdx.x & 7;
    int tid = threadIdx.x;
    float lgs = logit[(size_t)(E1 + J_NUM + c) * HEADS + h];
    // pass 1: max over 2048 job->crane logits (edge index i*32+c)
    float local = -3.4e38f;
    for (int i = tid; i < J_NUM; i += 256)
        local = fmaxf(local, logit[((size_t)i * C_NUM + c) * HEADS + h]);
    red[tid] = local; __syncthreads();
    for (int s = 128; s > 0; s >>= 1) { if (tid < s) red[tid] = fmaxf(red[tid], red[tid + s]); __syncthreads(); }
    float mx = fmaxf(red[0], lgs);
    __syncthreads();
    // pass 2: sum of exp
    local = 0.f;
    for (int i = tid; i < J_NUM; i += 256)
        local += __expf(logit[((size_t)i * C_NUM + c) * HEADS + h] - mx);
    red[tid] = local; __syncthreads();
    for (int s = 128; s > 0; s >>= 1) { if (tid < s) red[tid] += red[tid + s]; __syncthreads(); }
    float denom = red[0] + __expf(lgs - mx);
    __syncthreads();
    // pass 3: out[ch] = sum_i alpha_i * xl[job_i, h, ch]
    int ch = tid & 31, slice = tid >> 5;
    size_t fo = (size_t)h * CH + ch;
    float acc = 0.f;
    for (int i = slice; i < J_NUM; i += 8) {
        float a = __expf(logit[((size_t)i * C_NUM + c) * HEADS + h] - mx) / denom;
        acc += a * xl[(size_t)i * HID + fo];
    }
    red[tid] = acc; __syncthreads();
    if (tid < 32) {
        float tot = 0.f;
        #pragma unroll
        for (int s = 0; s < 8; ++s) tot += red[s * 32 + tid];
        tot += (__expf(lgs - mx) / denom) * xl[(size_t)(J_NUM + c) * HID + fo];
        size_t o = (size_t)(J_NUM + c) * HID + fo;
        xnew[o] = xold[o] + tot + bconv[fo];
    }
}

// ---------------- decoder layer 1: h1[e] = relu(A[src]+B[dst]+bd1), bf16 out ----------------
__global__ void h1_kernel(const float* __restrict__ A, const float* __restrict__ B,
                          const float* __restrict__ bd1,
                          const int* __restrict__ e2src, const int* __restrict__ e2dst,
                          unsigned short* __restrict__ h1) {
    int e = blockIdx.x, f = threadIdx.x;
    int s = e2src[e], d = e2dst[e];
    float v = A[(size_t)s * HID + f] + B[(size_t)d * HID + f] + bd1[f];
    v = v > 0.f ? v : 0.f;
    h1[(size_t)e * HID + f] = f32_to_bf16_bits(v);
}

// ---------------- final matvec: out[e] = h2[e,:]·wd3 + bd3 (wave per edge) ----------------
__global__ void matvec_kernel(const unsigned short* __restrict__ h2, const float* __restrict__ wd3,
                              const float* __restrict__ bd3, float* __restrict__ out) {
    int edge = blockIdx.x * 8 + (threadIdx.x >> 5);
    int lane = threadIdx.x & 31;
    const unsigned short* p = h2 + (size_t)edge * HID + lane * 8;
    float s = 0.f;
    #pragma unroll
    for (int q = 0; q < 8; ++q) s += bf16_bits_to_f32(p[q]) * wd3[lane * 8 + q];
    for (int o = 16; o >= 1; o >>= 1) s += __shfl_xor(s, o, 32);
    if (lane == 0) out[edge] = s + bd3[0];
}

// ---------------- host ----------------
extern "C" void kernel_launch(void* const* d_in, const int* in_sizes, int n_in,
                              void* d_out, int out_size, void* d_ws, size_t ws_size,
                              hipStream_t stream) {
    const float* nf1   = (const float*)d_in[0];
    const float* nf2   = (const float*)d_in[1];
    const float* Wenc1 = (const float*)d_in[2];
    const float* benc1 = (const float*)d_in[3];
    const float* Wenc2 = (const float*)d_in[4];
    const float* benc2 = (const float*)d_in[5];
    const float* Wl    = (const float*)d_in[6];
    const float* bl    = (const float*)d_in[7];
    const float* Wr    = (const float*)d_in[8];
    const float* br    = (const float*)d_in[9];
    const float* att   = (const float*)d_in[10];
    const float* bconv = (const float*)d_in[11];
    const float* Wd1   = (const float*)d_in[12];
    const float* bd1   = (const float*)d_in[13];
    const float* Wd2   = (const float*)d_in[14];
    const float* bd2   = (const float*)d_in[15];
    const float* Wd3   = (const float*)d_in[16];
    const float* bd3   = (const float*)d_in[17];
    const int*   e1    = (const int*)d_in[18];    // [2, E1]
    const int*   e2    = (const int*)d_in[19];    // [2, E2]
    float* out = (float*)d_out;

    // workspace carve-out (256B aligned)
    char* ws = (char*)d_ws;
    size_t off = 0;
    auto alloc = [&](size_t bytes) -> char* {
        char* p = ws + off;
        off = (off + bytes + 255) & ~(size_t)255;
        return p;
    };
    float*          xA    = (float*)alloc((size_t)NN * HID * 4);
    float*          xB    = (float*)alloc((size_t)NN * HID * 4);
    unsigned short* xbf   = (unsigned short*)alloc((size_t)NN * HID * 2);
    unsigned short* wT    = (unsigned short*)alloc((size_t)6 * HID * HID * 2); // 3 WlT + 3 WrT
    unsigned short* wd1tA = (unsigned short*)alloc((size_t)HID * HID * 2);
    unsigned short* wd1tB = (unsigned short*)alloc((size_t)HID * HID * 2);
    unsigned short* wd2t  = (unsigned short*)alloc((size_t)HID * HID * 2);
    float*          xl    = (float*)alloc((size_t)NN * HID * 4);
    float*          xr    = (float*)alloc((size_t)NN * HID * 4);
    float*          logit = (float*)alloc((size_t)ETOT * HEADS * 4);
    float*          Abuf  = (float*)alloc((size_t)NN * HID * 4);
    float*          Bbuf  = (float*)alloc((size_t)NN * HID * 4);
    unsigned short* h1    = (unsigned short*)alloc((size_t)E2 * HID * 2);
    unsigned short* h2    = (unsigned short*)alloc((size_t)E2 * HID * 2);
    (void)ws_size; (void)n_in; (void)in_sizes; (void)out_size;

    const int WE = HID * HID;                      // 65536 elems per 256x256 weight
    const int wgrid = (WE + 255) / 256;

    // ---- pre-transpose all GEMM weights to [N,K] bf16 ----
    for (int k = 0; k < 3; ++k) {
        transpose_bf16_kernel<<<wgrid, 256, 0, stream>>>(Wl + (size_t)k * WE, wT + (size_t)k * WE, HID, HID);
        transpose_bf16_kernel<<<wgrid, 256, 0, stream>>>(Wr + (size_t)k * WE, wT + (size_t)(3 + k) * WE, HID, HID);
    }
    transpose_bf16_kernel<<<wgrid, 256, 0, stream>>>(Wd1,                 wd1tA, HID, HID); // rows 0..255
    transpose_bf16_kernel<<<wgrid, 256, 0, stream>>>(Wd1 + (size_t)WE,    wd1tB, HID, HID); // rows 256..511
    transpose_bf16_kernel<<<wgrid, 256, 0, stream>>>(Wd2,                 wd2t,  HID, HID);

    // ---- encoder ----
    encoder_kernel<<<NN, HID, 0, stream>>>(nf1, nf2, Wenc1, benc1, Wenc2, benc2, xA);

    // ---- 3 residual GATv2 layers ----
    float* xcur = xA;
    float* xnxt = xB;
    const dim3 gNode((NN + 127) / 128, HID / 64);
    const int lgrid = (ETOT * HEADS + 255) / 256;
    for (int k = 0; k < 3; ++k) {
        to_bf16_kernel<<<(NN * HID + 255) / 256, 256, 0, stream>>>(xcur, xbf, NN * HID);
        gemm_bf16_wmma<false, false><<<gNode, 256, 0, stream>>>(xbf, wT + (size_t)k * WE,       bl + k * HID, xl, NN, HID, HID);
        gemm_bf16_wmma<false, false><<<gNode, 256, 0, stream>>>(xbf, wT + (size_t)(3 + k) * WE, br + k * HID, xr, NN, HID, HID);
        edge_logit_kernel<<<lgrid, 256, 0, stream>>>(xl, xr, att + (size_t)k * HID, e1, e1 + E1, logit);
        agg_job_kernel<<<J_NUM, 256, 0, stream>>>(xcur, xl, logit, bconv + k * HID, xnxt);
        agg_crane_kernel<<<C_NUM * HEADS, 256, 0, stream>>>(xcur, xl, logit, bconv + k * HID, xnxt);
        float* t = xcur; xcur = xnxt; xnxt = t;
    }

    // ---- decoder ----
    to_bf16_kernel<<<(NN * HID + 255) / 256, 256, 0, stream>>>(xcur, xbf, NN * HID);
    gemm_bf16_wmma<false, false><<<gNode, 256, 0, stream>>>(xbf, wd1tA, nullptr, Abuf, NN, HID, HID);
    gemm_bf16_wmma<false, false><<<gNode, 256, 0, stream>>>(xbf, wd1tB, nullptr, Bbuf, NN, HID, HID);
    h1_kernel<<<E2, HID, 0, stream>>>(Abuf, Bbuf, bd1, e2, e2 + E2, h1);
    const dim3 gH2((E2 + 127) / 128, HID / 64);
    gemm_bf16_wmma<true, true><<<gH2, 256, 0, stream>>>(h1, wd2t, bd2, h2, E2, HID, HID);
    matvec_kernel<<<E2 / 8, 256, 0, stream>>>(h2, Wd3, bd3, out);
}
